// CrossAttention_15083925143658
// MI455X (gfx1250) — matrix operand order repaired and checked by reading, not compiled, not run on hardware
//
#include <hip/hip_runtime.h>
#include <hip/hip_bf16.h>

typedef __attribute__((ext_vector_type(16))) _Float16 v16h;
typedef __attribute__((ext_vector_type(8)))  float    v8f;
typedef __attribute__((ext_vector_type(8)))  _Float16 half8;
typedef __attribute__((ext_vector_type(4)))  float    f32x4;

namespace {
constexpr int BATCH = 32;
constexpr int NLAT  = 64;
constexpr int NPAT  = 4096;
constexpr int DIM   = 192;   // DIM_LAT = DIM_PAT = DIM_ATTN
constexpr int NH    = 6;
constexpr int HD    = 32;
constexpr float SCALE = 0.17677669529663687f;  // 1/sqrt(HD)
}

__device__ __forceinline__ v8f wmma_f16(v16h a, v16h b, v8f c) {
  return __builtin_amdgcn_wmma_f32_16x16x32_f16(
      /*neg_a=*/false, a, /*neg_b=*/false, b,
      /*c_mod=*/(short)0, c, /*reuse_a=*/false, /*reuse_b=*/false);
}

// A fragment: 16(M) x 32(K) f16 from row-major LDS [row][k], stride ld.
__device__ __forceinline__ v16h frag_a(const _Float16* p, int ld) {
  const int lane = threadIdx.x & 31;
  const _Float16* r = p + (lane & 15) * ld + ((lane >> 4) << 3);
  half8 lo = *(const half8*)r;
  half8 hi = *(const half8*)(r + 16);
  return __builtin_shufflevector(lo, hi, 0, 1, 2, 3, 4, 5, 6, 7,
                                 8, 9, 10, 11, 12, 13, 14, 15);
}

// B fragment: 32(K) x 16(N) f16 from TRANSPOSED storage [n][k], stride ld.
__device__ __forceinline__ v16h frag_bt(const _Float16* p, int ld) {
  const int lane = threadIdx.x & 31;
  const _Float16* cp = p + (lane & 15) * ld + ((lane >> 4) << 3);
  half8 lo = *(const half8*)cp;
  half8 hi = *(const half8*)(cp + 16);
  return __builtin_shufflevector(lo, hi, 0, 1, 2, 3, 4, 5, 6, 7,
                                 8, 9, 10, 11, 12, 13, 14, 15);
}

// C/D fragment (16x16 f32) -> LDS f32 row-major, stride ld (floats).
__device__ __forceinline__ void store_c_lds(float* p, int ld, v8f c) {
  const int lane = threadIdx.x & 31;
  float* r = p + ((lane >> 4) << 3) * ld + (lane & 15);
#pragma unroll
  for (int i = 0; i < 8; ++i) r[i * ld] = c[i];
}

// C/D fragment -> LDS f16 row-major, stride ld (halves).
__device__ __forceinline__ void store_c_lds_f16(_Float16* p, int ld, v8f c) {
  const int lane = threadIdx.x & 31;
  _Float16* r = p + ((lane >> 4) << 3) * ld + (lane & 15);
#pragma unroll
  for (int i = 0; i < 8; ++i) r[i * ld] = (_Float16)c[i];
}

// ---------------------------------------------------------------------------
// Kernel 0: one-time weight prep.  W f32 [192][N] -> WT f16 [N][192].
// ---------------------------------------------------------------------------
__global__ __launch_bounds__(192) void k_prep_wt(
    const float* __restrict__ W, _Float16* __restrict__ WT, int N) {
  const int k = threadIdx.x;   // 0..191
  const int n = blockIdx.x;    // 0..N-1
  WT[(size_t)n * DIM + k] = (_Float16)W[(size_t)k * N + n];
}

// ---------------------------------------------------------------------------
// Kernel 1: rv projection.  X[M,192] (f32) @ W[192,384] -> f16
//   Full-width tile: 64 rows x 384 cols per block (X read exactly once).
//   r stored row-major  [B, H, Ntok, HD]
//   v stored TRANSPOSED [B, H, HD, Ntok]  (C-fragment contiguous this way)
// ---------------------------------------------------------------------------
__global__ __launch_bounds__(256) void k_proj_rv(
    const float* __restrict__ X, const _Float16* __restrict__ WT,  // [384][192]
    _Float16* __restrict__ rbuf, _Float16* __restrict__ vbuf, int tokShift) {
  __shared__ _Float16 sA[64][40];     // x chunk [m][k]
  __shared__ _Float16 sBT[384][40];   // WT chunk [n][k]
  __shared__ _Float16 sOut[64][200];  // staged f16 r-half (192 cols)
  const int t = threadIdx.x;
  const int wave = t >> 5;
  const int m0 = blockIdx.x * 64;
  const int tr = wave >> 1;
  const int par = wave & 1;           // tc = par + 2*j
  v8f acc[12] = {};
  for (int k0 = 0; k0 < DIM; k0 += 32) {
    {  // A chunk [64,32] f32 -> f16
      const int r = t >> 2, cs2 = (t & 3) * 8;
      const float* src = X + (size_t)(m0 + r) * DIM + k0 + cs2;
      f32x4 a0 = *(const f32x4*)src;
      f32x4 a1 = *(const f32x4*)(src + 4);
      half8 hv;
#pragma unroll
      for (int i = 0; i < 4; ++i) { hv[i] = (_Float16)a0[i]; hv[4 + i] = (_Float16)a1[i]; }
      *(half8*)&sA[r][cs2] = hv;
    }
    {  // WT chunk [384][32]: straight f16 vector copies
      const int k8 = (t & 3) * 8;
#pragma unroll
      for (int it = 0; it < 6; ++it) {
        const int n = it * 64 + (t >> 2);
        *(half8*)&sBT[n][k8] = *(const half8*)&WT[(size_t)n * DIM + k0 + k8];
      }
    }
    __syncthreads();
    v16h a = frag_a(&sA[tr * 16][0], 40);
#pragma unroll
    for (int j = 0; j < 12; ++j) {
      v16h bf = frag_bt(&sBT[(par + 2 * j) * 16][0], 40);
      acc[j] = wmma_f16(a, bf, acc[j]);
    }
    __syncthreads();
  }
  const int lane = t & 31;
  const int colp = lane & 15;
  const int rb = (lane >> 4) << 3;
  {  // v half (j = 6..11, cols 192..383): direct transposed b128 stores
    const int m = m0 + tr * 16 + rb;
    const int bb = m >> tokShift;
    const int nn = m & ((1 << tokShift) - 1);
#pragma unroll
    for (int j = 6; j < 12; ++j) {
      const int cm = (par + 2 * j) * 16 + colp - DIM;  // 0..191
      const int hh = cm >> 5;
      const int d  = cm & 31;
      half8 hv;
#pragma unroll
      for (int i = 0; i < 8; ++i) hv[i] = (_Float16)acc[j][i];
      *(half8*)&vbuf[(((size_t)bb * NH + hh) * HD + d) * ((size_t)1 << tokShift) + nn] = hv;
    }
  }
  // r half (j = 0..5, cols 0..191): stage + cooperative coalesced store
#pragma unroll
  for (int j = 0; j < 6; ++j)
    store_c_lds_f16(&sOut[tr * 16][(par + 2 * j) * 16], 200, acc[j]);
  __syncthreads();
  {
    const int r = t >> 2;
    const int m = m0 + r;
    const int bb = m >> tokShift;
    const int nn = m & ((1 << tokShift) - 1);
#pragma unroll
    for (int q = 0; q < 3; ++q) {
      const int c = (t & 3) * 48 + q * 16;  // 16-chunk stays inside one head
      const int hh = c >> 5;
      const int d0 = c & 31;  // 0 or 16
      _Float16* dst = rbuf +
          (((size_t)bb * NH + hh) * ((size_t)1 << tokShift) + nn) * HD + d0;
      *(half8*)dst       = *(const half8*)&sOut[r][c];
      *(half8*)(dst + 8) = *(const half8*)&sOut[r][c + 8];
    }
  }
}

// ---------------------------------------------------------------------------
// Kernel 2: fused bidirectional attention, one workgroup per (b,h).
// ---------------------------------------------------------------------------
__global__ __launch_bounds__(256) void k_attn(
    const _Float16* __restrict__ rlat, const _Float16* __restrict__ vlatT,
    const _Float16* __restrict__ rpat, const _Float16* __restrict__ vpatT,
    _Float16* __restrict__ aolat, _Float16* __restrict__ aopat) {
  __shared__ _Float16 s_vlatT[32][72];   // [d][l]  (B^T for out_pat)
  __shared__ _Float16 s_rpat[64][40];    // [p][d]  (B^T for scores; reused to stage out_pat)
  __shared__ _Float16 s_vpatT[32][72];   // [d][p]  (B^T for out_lat)
  __shared__ float    s_S[64][68];
  __shared__ _Float16 s_attn[64][72];    // row-softmax  [l][p]
  __shared__ _Float16 s_attnT[64][72];   // col-softmax  [p][l]
  __shared__ float s_pm[4][64];
  __shared__ float s_ps[4][64];
  __shared__ float s_m[64], s_s[64], s_cm[64], s_cs[64];

  const int t = threadIdx.x;
  const int wave = t >> 5;
  const int bh = blockIdx.x;
  const int b = bh / NH;
  const int h = bh - b * NH;
  const int row = t & 63;
  const int cg  = t >> 6;
  const int otr = wave >> 1;
  const int odc = wave & 1;
  const int lane = t & 31;

  {  // v_lat: straight transposed-layout copy
    const _Float16* gv = vlatT + (size_t)bh * HD * NLAT;
    const int d = t >> 3, l8 = (t & 7) * 8;
    *(half8*)&s_vlatT[d][l8] = *(const half8*)&gv[d * NLAT + l8];
  }
  // r_lat A-fragment straight from global (wave-private rows), f16 scale
  v16h a_rlat;
  {
    const _Float16* gr = rlat + (size_t)bh * NLAT * HD;
    const int l = otr * 16 + (lane & 15);
    const int kg = (lane >> 4) << 3;
    half8 lo = *(const half8*)&gr[l * HD + kg];
    half8 hi = *(const half8*)&gr[l * HD + kg + 16];
    const _Float16 hs = (_Float16)SCALE;
#pragma unroll
    for (int i = 0; i < 8; ++i) { lo[i] = lo[i] * hs; hi[i] = hi[i] * hs; }
    a_rlat = __builtin_shufflevector(lo, hi, 0, 1, 2, 3, 4, 5, 6, 7,
                                     8, 9, 10, 11, 12, 13, 14, 15);
  }
  __syncthreads();

  v16h b_vlat[2];
#pragma unroll
  for (int kc = 0; kc < 2; ++kc)
    b_vlat[kc] = frag_bt(&s_vlatT[odc * 16][kc * 32], 72);

  const int stc0 = (wave & 1) * 2;
  auto computeS = [&]() {
#pragma unroll
    for (int j = 0; j < 2; ++j) {
      v16h bf = frag_bt(&s_rpat[(stc0 + j) * 16][0], 40);
      v8f cs = {};
      cs = wmma_f16(a_rlat, bf, cs);
      store_c_lds(&s_S[otr * 16][(stc0 + j) * 16], 68, cs);
    }
  };

  // -------- pass 1: exact row (patch-axis) softmax stats --------
  float m_run = -3.0e38f, s_run = 0.0f;
  for (int p0 = 0; p0 < NPAT; p0 += 64) {
    {
      const _Float16* g = rpat + ((size_t)bh * NPAT + p0) * HD;
      const int p = t >> 2, d = (t & 3) * 8;
      *(half8*)&s_rpat[p][d] = *(const half8*)&g[p * HD + d];
      if (p0 + 64 < NPAT)
        __builtin_prefetch((const char*)(g + 64 * HD) + t * 16, 0, 0);
    }
    __syncthreads();
    computeS();
    __syncthreads();
    float sv[16];
#pragma unroll
    for (int j = 0; j < 4; ++j) {
      f32x4 q = *(const f32x4*)&s_S[row][cg * 16 + j * 4];
#pragma unroll
      for (int i = 0; i < 4; ++i) sv[j * 4 + i] = q[i];
    }
    float tmax = sv[0];
#pragma unroll
    for (int i = 1; i < 16; ++i) tmax = fmaxf(tmax, sv[i]);
    float tsum = 0.0f;
#pragma unroll
    for (int i = 0; i < 16; ++i) tsum += __expf(sv[i] - tmax);
    const float nm = fmaxf(m_run, tmax);
    s_run = s_run * __expf(m_run - nm) + tsum * __expf(tmax - nm);
    m_run = nm;
    __syncthreads();
  }
  s_pm[cg][row] = m_run;
  s_ps[cg][row] = s_run;
  __syncthreads();
  if (t < 64) {
    float m = s_pm[0][t];
#pragma unroll
    for (int i = 1; i < 4; ++i) m = fmaxf(m, s_pm[i][t]);
    float s = 0.0f;
#pragma unroll
    for (int i = 0; i < 4; ++i) s += s_ps[i][t] * __expf(s_pm[i][t] - m);
    s_m[t] = m;
    s_s[t] = 1.0f / s;  // reciprocal
  }
  __syncthreads();

  // -------- pass 2: softmaxes + both attention applies --------
  v8f c_lat = {};
  for (int p0 = 0; p0 < NPAT; p0 += 64) {
    {
      const _Float16* gr = rpat + ((size_t)bh * NPAT + p0) * HD;
      const _Float16* gv = vpatT + (size_t)bh * HD * NPAT;
      const int p = t >> 2, d = (t & 3) * 8;
      *(half8*)&s_rpat[p][d] = *(const half8*)&gr[p * HD + d];
      const int dt = t >> 3, p8 = (t & 7) * 8;
      *(half8*)&s_vpatT[dt][p8] =
          *(const half8*)&gv[(size_t)dt * NPAT + p0 + p8];
      if (p0 + 64 < NPAT) {
        __builtin_prefetch((const char*)(gr + 64 * HD) + t * 16, 0, 0);
        __builtin_prefetch(
            (const char*)(gv + (size_t)dt * NPAT + p0 + 64) + (t & 7) * 16, 0, 0);
      }
    }
    __syncthreads();
    computeS();
    __syncthreads();
    // column (latent-axis) softmax: cache column quarter in registers
    float sv[16];
    {
      const int col = t & 63;
      const int rg = t >> 6;
#pragma unroll
      for (int i = 0; i < 16; ++i) sv[i] = s_S[rg * 16 + i][col];
      float tmax = sv[0];
#pragma unroll
      for (int i = 1; i < 16; ++i) tmax = fmaxf(tmax, sv[i]);
      float tsum = 0.0f;
#pragma unroll
      for (int i = 0; i < 16; ++i) tsum += __expf(sv[i] - tmax);
      s_pm[rg][col] = tmax;
      s_ps[rg][col] = tsum;
    }
    __syncthreads();
    if (t < 64) {
      float m = s_pm[0][t];
#pragma unroll
      for (int i = 1; i < 4; ++i) m = fmaxf(m, s_pm[i][t]);
      float s = 0.0f;
#pragma unroll
      for (int i = 0; i < 4; ++i) s += s_ps[i][t] * __expf(s_pm[i][t] - m);
      s_cm[t] = m;
      s_cs[t] = 1.0f / s;
    }
    __syncthreads();
    {  // attnT from registers: vector stores
      const int col = t & 63;
      const int rg = t >> 6;
      const float cm = s_cm[col], cs = s_cs[col];
      half8 h0, h1;
#pragma unroll
      for (int i = 0; i < 8; ++i) {
        h0[i] = (_Float16)(__expf(sv[i] - cm) * cs);
        h1[i] = (_Float16)(__expf(sv[8 + i] - cm) * cs);
      }
      *(half8*)&s_attnT[col][rg * 16]     = h0;
      *(half8*)&s_attnT[col][rg * 16 + 8] = h1;
    }
    {  // attn (row softmax): vector row reads + vector stores
      const float rm = s_m[row];
      const float rinv = s_s[row];
      half8 h0, h1;
#pragma unroll
      for (int j = 0; j < 2; ++j) {
        f32x4 q0 = *(const f32x4*)&s_S[row][cg * 16 + j * 8];
        f32x4 q1 = *(const f32x4*)&s_S[row][cg * 16 + j * 8 + 4];
        half8& hd = j ? h1 : h0;
#pragma unroll
        for (int i = 0; i < 4; ++i) {
          hd[i]     = (_Float16)(__expf(q0[i] - rm) * rinv);
          hd[4 + i] = (_Float16)(__expf(q1[i] - rm) * rinv);
        }
      }
      *(half8*)&s_attn[row][cg * 16]     = h0;
      *(half8*)&s_attn[row][cg * 16 + 8] = h1;
    }
    __syncthreads();
    // out_lat[64,32] += attn[64,64] @ v_pat[64,32]
#pragma unroll
    for (int kc = 0; kc < 2; ++kc) {
      v16h a = frag_a(&s_attn[otr * 16][kc * 32], 72);
      v16h bf = frag_bt(&s_vpatT[odc * 16][kc * 32], 72);
      c_lat = wmma_f16(a, bf, c_lat);
    }
    // out_pat[64,32] = attnT[64,64] @ v_lat[64,32]
    v8f c_pat = {};
#pragma unroll
    for (int kc = 0; kc < 2; ++kc) {
      v16h a = frag_a(&s_attnT[otr * 16][kc * 32], 72);
      c_pat = wmma_f16(a, b_vlat[kc], c_pat);
    }
    store_c_lds_f16(&s_rpat[otr * 16][odc * 16], 40, c_pat);
    __syncthreads();
    {
      const int p = t >> 2, d = (t & 3) * 8;
      _Float16* dst = aopat + ((size_t)b * NPAT + p0 + p) * DIM + h * HD + d;
      *(half8*)dst = *(const half8*)&s_rpat[p][d];
    }
    __syncthreads();
  }
  {  // final out_lat store (once)
    const int colp = lane & 15;
    const int rb = (lane >> 4) << 3;
#pragma unroll
    for (int i = 0; i < 8; ++i) {
      const int l = otr * 16 + rb + i;
      const int d = odc * 16 + colp;
      aolat[((size_t)b * NLAT + l) * DIM + h * HD + d] = (_Float16)c_lat[i];
    }
  }
}

// ---------------------------------------------------------------------------
// Kernel 3: output projection.  A[M,192] (f16) @ WT[192][192] (f16) + bias.
// Full-width tile: 64 rows x 192 cols (A read exactly once).
// ---------------------------------------------------------------------------
__global__ __launch_bounds__(256) void k_proj_out(
    const _Float16* __restrict__ A, const _Float16* __restrict__ WT,
    const float* __restrict__ bias, float* __restrict__ Out) {
  __shared__ _Float16 sA[64][40];
  __shared__ _Float16 sBT[192][40];
  const int t = threadIdx.x;
  const int wave = t >> 5;
  const int m0 = blockIdx.x * 64;
  const int tr = wave >> 1;
  const int par = wave & 1;  // tc = par + 2*j
  v8f acc[6] = {};
  for (int k0 = 0; k0 < DIM; k0 += 32) {
    {
      const int r = t >> 2, cs2 = (t & 3) * 8;
      *(half8*)&sA[r][cs2] = *(const half8*)&A[(size_t)(m0 + r) * DIM + k0 + cs2];
    }
    {
      const int k8 = (t & 3) * 8;
#pragma unroll
      for (int it = 0; it < 3; ++it) {
        const int n = it * 64 + (t >> 2);
        *(half8*)&sBT[n][k8] = *(const half8*)&WT[(size_t)n * DIM + k0 + k8];
      }
    }
    __syncthreads();
    v16h a = frag_a(&sA[tr * 16][0], 40);
#pragma unroll
    for (int j = 0; j < 6; ++j) {
      v16h bf = frag_bt(&sBT[(par + 2 * j) * 16][0], 40);
      acc[j] = wmma_f16(a, bf, acc[j]);
    }
    __syncthreads();
  }
  const int lane = t & 31;
  const int colp = lane & 15;
  const int rb = (lane >> 4) << 3;
#pragma unroll
  for (int j = 0; j < 6; ++j) {  // lanes cover consecutive cols: coalesced b32
#pragma unroll
    for (int i = 0; i < 8; ++i) {
      const int m = m0 + tr * 16 + rb + i;
      const int c = (par + 2 * j) * 16 + colp;
      Out[(size_t)m * DIM + c] = acc[j][i] + bias[c];
    }
  }
}

extern "C" void kernel_launch(void* const* d_in, const int* in_sizes, int n_in,
                              void* d_out, int out_size, void* d_ws, size_t ws_size,
                              hipStream_t stream) {
  (void)in_sizes; (void)n_in; (void)out_size; (void)ws_size;
  const float* x_lat = (const float*)d_in[0];
  const float* x_pat = (const float*)d_in[1];
  const float* W_rv_lat = (const float*)d_in[2];
  const float* W_rv_pat = (const float*)d_in[3];
  const float* W_pl = (const float*)d_in[4];
  const float* b_pl = (const float*)d_in[5];
  const float* W_pp = (const float*)d_in[6];
  const float* b_pp = (const float*)d_in[7];
  float* out = (float*)d_out;

  char* ws = (char*)d_ws;
  size_t off = 0;
  auto take = [&](size_t bytes) -> char* {
    char* p = ws + off;
    off += (bytes + 255) & ~(size_t)255;
    return p;
  };
  _Float16* r_lat  = (_Float16*)take((size_t)BATCH * NH * NLAT * HD * 2);
  _Float16* v_lat  = (_Float16*)take((size_t)BATCH * NH * NLAT * HD * 2);  // [B,H,HD,NLAT]
  _Float16* r_pat  = (_Float16*)take((size_t)BATCH * NH * NPAT * HD * 2);
  _Float16* v_pat  = (_Float16*)take((size_t)BATCH * NH * NPAT * HD * 2);  // [B,H,HD,NPAT]
  _Float16* ao_lat = (_Float16*)take((size_t)BATCH * NLAT * DIM * 2);
  _Float16* ao_pat = (_Float16*)take((size_t)BATCH * NPAT * DIM * 2);
  _Float16* wt_rvl = (_Float16*)take((size_t)(2 * DIM) * DIM * 2);
  _Float16* wt_rvp = (_Float16*)take((size_t)(2 * DIM) * DIM * 2);
  _Float16* wt_pl  = (_Float16*)take((size_t)DIM * DIM * 2);
  _Float16* wt_pp  = (_Float16*)take((size_t)DIM * DIM * 2);
  // total workspace: ~153 MB + 0.5 MB weights

  // one-time weight transpose+convert
  k_prep_wt<<<dim3(2 * DIM), 192, 0, stream>>>(W_rv_lat, wt_rvl, 2 * DIM);
  k_prep_wt<<<dim3(2 * DIM), 192, 0, stream>>>(W_rv_pat, wt_rvp, 2 * DIM);
  k_prep_wt<<<dim3(DIM), 192, 0, stream>>>(W_pl, wt_pl, DIM);
  k_prep_wt<<<dim3(DIM), 192, 0, stream>>>(W_pp, wt_pp, DIM);

  k_proj_rv<<<dim3((BATCH * NLAT) / 64), 256, 0, stream>>>(
      x_lat, wt_rvl, r_lat, v_lat, 6);    // Ntok = 64
  k_proj_rv<<<dim3((BATCH * NPAT) / 64), 256, 0, stream>>>(
      x_pat, wt_rvp, r_pat, v_pat, 12);   // Ntok = 4096
  k_attn<<<dim3(BATCH * NH), 256, 0, stream>>>(r_lat, v_lat, r_pat, v_pat,
                                               ao_lat, ao_pat);
  k_proj_out<<<dim3((BATCH * NLAT) / 64), 256, 0, stream>>>(
      ao_lat, wt_pl, b_pl, out);
  k_proj_out<<<dim3((BATCH * NPAT) / 64), 256, 0, stream>>>(
      ao_pat, wt_pp, b_pp, out + (size_t)BATCH * NLAT * DIM);
}
